// RecurrentDQN_22746146800069
// MI455X (gfx1250) — compile-verified, matrix operand order checked by reading, not run on hardware
//
#include <hip/hip_runtime.h>
#include <hip/hip_bf16.h>
#include <math.h>

typedef __bf16 bf16;
typedef bf16  bf16x4  __attribute__((ext_vector_type(4)));
typedef bf16  bf16x8  __attribute__((ext_vector_type(8)));
typedef bf16  v16bf   __attribute__((ext_vector_type(16)));
typedef float v8f     __attribute__((ext_vector_type(8)));

constexpr int T_   = 512;
constexpr int B_   = 64;
constexpr int OBS_ = 512;
constexpr int H_   = 1024;
constexpr int A_   = 18;
constexpr int TB_  = T_ * B_;     // 32768 flattened (t,b) rows
constexpr int G_   = 4 * H_;      // 4096 gate columns

constexpr int KC  = 32;           // K chunk per WMMA (bf16 16x16x32)
constexpr int LDT = KC + 8;       // padded LDS row stride (halves), keeps 16B align
constexpr int NK  = H_ / KC;      // 32 K-chunks in the recurrent GEMM

// ---------------------------------------------------------------------------
// Fragment load helpers (layouts per CDNA5 ISA 7.12.2, wave32)
// ---------------------------------------------------------------------------
__device__ __forceinline__ v16bf frag_ld(const bf16* base) {
  int lane = threadIdx.x & 31;
  int r = lane & 15;
  int h = lane >> 4;
  const bf16* p = base + r * LDT + h * 8;
  bf16x8 lo = *(const bf16x8*)(p);
  bf16x8 hi = *(const bf16x8*)(p + 16);
  return __builtin_shufflevector(lo, hi, 0,1,2,3,4,5,6,7,8,9,10,11,12,13,14,15);
}

__device__ __forceinline__ v8f wmma_bf16(v16bf a, v16bf b, v8f c) {
  return __builtin_amdgcn_wmma_f32_16x16x32_bf16(false, a, false, b, (short)0, c,
                                                 false, false);
}

__device__ __forceinline__ float sigmoidf_(float x) {
  return 1.0f / (1.0f + __expf(-x));
}

// ---------------------------------------------------------------------------
// gfx1250 async global->LDS copy (GLOBAL_LOAD_ASYNC_TO_LDS_B128, ASYNCcnt)
// LDS offset = low 32 bits of the generic shared pointer (ISA 10.2 aperture).
// ---------------------------------------------------------------------------
__device__ __forceinline__ void async_ld_b128(void* lds, const void* g) {
  unsigned lo = (unsigned)(size_t)lds;
  asm volatile("global_load_async_to_lds_b128 %0, %1, off"
               :: "v"(lo), "v"(g) : "memory");
}
__device__ __forceinline__ void wait_async_le5() {
  asm volatile("s_wait_asynccnt 5" ::: "memory");
}
__device__ __forceinline__ void wait_async_0() {
  asm volatile("s_wait_asynccnt 0" ::: "memory");
}

// ---------------------------------------------------------------------------
// init: zero h ping buffer (bf16) and the device barrier counter
// ---------------------------------------------------------------------------
__global__ void init_kernel(bf16* __restrict__ h0, unsigned* __restrict__ bar) {
  int i = blockIdx.x * blockDim.x + threadIdx.x;
  if (i < B_ * H_) h0[i] = (bf16)0.0f;
  if (i == 0) *bar = 0u;
}

// ---------------------------------------------------------------------------
// convert W_hh (4H x H fp32) -> bf16 once; stays hot in 192MB L2 for all steps
// ---------------------------------------------------------------------------
__global__ void cvt_whh_kernel(const float* __restrict__ w, bf16* __restrict__ o) {
  int i = blockIdx.x * blockDim.x + threadIdx.x; // over G_*H_/4
  if (i < (G_ * H_) / 4) {
    float4 v = *(const float4*)(w + (size_t)i * 4);
    bf16x4 b = { (bf16)v.x, (bf16)v.y, (bf16)v.z, (bf16)v.w };
    *(bf16x4*)(o + (size_t)i * 4) = b;
  }
}

// ---------------------------------------------------------------------------
// encoder: enc[TB,H] = relu(obs[TB,OBS] @ Wenc[H,OBS]^T + benc) -> bf16
// ---------------------------------------------------------------------------
__global__ __launch_bounds__(256) void enc_kernel(
    const float* __restrict__ obs, const float* __restrict__ Wenc,
    const float* __restrict__ benc, bf16* __restrict__ enc_out) {
  __shared__ __align__(16) bf16 As[128 * LDT];
  __shared__ __align__(16) bf16 Bs[128 * LDT];
  int tid  = threadIdx.x;
  int wave = tid >> 5;
  int row0 = blockIdx.x * 128;
  int col0 = blockIdx.y * 128;
  int wr = (wave >> 1) * 32;
  int wc = (wave & 1) * 64;

  v8f acc[2][4];
  for (int i = 0; i < 2; ++i)
    for (int j = 0; j < 4; ++j) acc[i][j] = (v8f)(0.0f);

  for (int k0 = 0; k0 < OBS_; k0 += KC) {
    __syncthreads();
    for (int i = tid; i < 128 * (KC / 4); i += 256) {   // A: fp32 -> bf16
      int r = i >> 3, c = (i & 7) * 4;
      float4 v = *(const float4*)(obs + (size_t)(row0 + r) * OBS_ + k0 + c);
      bf16x4 b = { (bf16)v.x, (bf16)v.y, (bf16)v.z, (bf16)v.w };
      *(bf16x4*)(&As[r * LDT + c]) = b;
    }
    for (int i = tid; i < 128 * (KC / 4); i += 256) {   // B: Wenc rows
      int r = i >> 3, c = (i & 7) * 4;
      float4 v = *(const float4*)(Wenc + (size_t)(col0 + r) * OBS_ + k0 + c);
      bf16x4 b = { (bf16)v.x, (bf16)v.y, (bf16)v.z, (bf16)v.w };
      *(bf16x4*)(&Bs[r * LDT + c]) = b;
    }
    __syncthreads();
    v16bf a0 = frag_ld(&As[(wr +  0) * LDT]);
    v16bf a1 = frag_ld(&As[(wr + 16) * LDT]);
    for (int ct = 0; ct < 4; ++ct) {
      v16bf b = frag_ld(&Bs[(wc + ct * 16) * LDT]);
      acc[0][ct] = wmma_bf16(a0, b, acc[0][ct]);
      acc[1][ct] = wmma_bf16(a1, b, acc[1][ct]);
    }
  }

  int lane = tid & 31, cn = lane & 15, mo = (lane >> 4) * 8;
  for (int rt = 0; rt < 2; ++rt)
    for (int ct = 0; ct < 4; ++ct) {
      int n = col0 + wc + ct * 16 + cn;
      float bias = benc[n];
      for (int r = 0; r < 8; ++r) {
        int m = row0 + wr + rt * 16 + mo + r;
        float v = acc[rt][ct][r] + bias;
        v = v > 0.0f ? v : 0.0f;
        enc_out[(size_t)m * H_ + n] = (bf16)v;
      }
    }
}

// ---------------------------------------------------------------------------
// x-gates: xg[TB,4H] = enc[TB,H] @ Wih[4H,H]^T + (bih + bhh)  -> bf16
// ---------------------------------------------------------------------------
__global__ __launch_bounds__(256) void xg_kernel(
    const bf16* __restrict__ enc, const float* __restrict__ Wih,
    const float* __restrict__ bih, const float* __restrict__ bhh,
    bf16* __restrict__ xg_out) {
  __shared__ __align__(16) bf16 As[128 * LDT];
  __shared__ __align__(16) bf16 Bs[128 * LDT];
  int tid  = threadIdx.x;
  int wave = tid >> 5;
  int row0 = blockIdx.x * 128;
  int col0 = blockIdx.y * 128;
  int wr = (wave >> 1) * 32;
  int wc = (wave & 1) * 64;

  v8f acc[2][4];
  for (int i = 0; i < 2; ++i)
    for (int j = 0; j < 4; ++j) acc[i][j] = (v8f)(0.0f);

  for (int k0 = 0; k0 < H_; k0 += KC) {
    __syncthreads();
    for (int i = tid; i < 128 * (KC / 8); i += 256) {   // A already bf16
      int r = i >> 2, c = (i & 3) * 8;
      *(bf16x8*)(&As[r * LDT + c]) =
          *(const bf16x8*)(enc + (size_t)(row0 + r) * H_ + k0 + c);
    }
    for (int i = tid; i < 128 * (KC / 4); i += 256) {   // B: Wih fp32 rows
      int r = i >> 3, c = (i & 7) * 4;
      float4 v = *(const float4*)(Wih + (size_t)(col0 + r) * H_ + k0 + c);
      bf16x4 b = { (bf16)v.x, (bf16)v.y, (bf16)v.z, (bf16)v.w };
      *(bf16x4*)(&Bs[r * LDT + c]) = b;
    }
    __syncthreads();
    v16bf a0 = frag_ld(&As[(wr +  0) * LDT]);
    v16bf a1 = frag_ld(&As[(wr + 16) * LDT]);
    for (int ct = 0; ct < 4; ++ct) {
      v16bf b = frag_ld(&Bs[(wc + ct * 16) * LDT]);
      acc[0][ct] = wmma_bf16(a0, b, acc[0][ct]);
      acc[1][ct] = wmma_bf16(a1, b, acc[1][ct]);
    }
  }

  int lane = tid & 31, cn = lane & 15, mo = (lane >> 4) * 8;
  for (int rt = 0; rt < 2; ++rt)
    for (int ct = 0; ct < 4; ++ct) {
      int n = col0 + wc + ct * 16 + cn;
      float bias = bih[n] + bhh[n];
      for (int r = 0; r < 8; ++r) {
        int m = row0 + wr + rt * 16 + mo + r;
        xg_out[(size_t)m * G_ + n] = (bf16)(acc[rt][ct][r] + bias);
      }
    }
}

// ---------------------------------------------------------------------------
// persistent LSTM scan: 16 WGs, each owns 64 batch x 64 h-cols for the whole
// scan (c state lives in VGPRs). Per step: 4-gate WMMA GEMM with async-to-LDS
// double-buffered staging, fused cell epilogue, device-scope barrier.
// ---------------------------------------------------------------------------
__global__ __launch_bounds__(256) void lstm_scan_kernel(
    bf16* __restrict__ hbu0, bf16* __restrict__ hbu1,
    const bf16* __restrict__ xg, const bf16* __restrict__ Whh,
    const float* __restrict__ mask, bf16* __restrict__ latent,
    float* __restrict__ outH, float* __restrict__ outC,
    unsigned* __restrict__ bar) {
  __shared__ __align__(16) bf16 As0[64 * LDT];
  __shared__ __align__(16) bf16 As1[64 * LDT];
  __shared__ __align__(16) bf16 Bs0[256 * LDT];   // 4 gates x 64 cols
  __shared__ __align__(16) bf16 Bs1[256 * LDT];
  int tid  = threadIdx.x;
  int wave = tid >> 5;
  int hc0  = blockIdx.x * 64;
  int rt   = wave >> 1;            // row tile 0..3 -> rows rt*16
  int cth  = (wave & 1) * 32;      // col offset within 64
  int lane = tid & 31, cn = lane & 15, mo = (lane >> 4) * 8;

  // per-thread cell state: c[(j,r)] for the (m,n) pairs this thread owns
  float creg[2][8];
  for (int j = 0; j < 2; ++j)
    for (int r = 0; r < 8; ++r) creg[j][r] = 0.0f;

  // A staging indices (64x32 bf16, one b128 per thread)
  int ar = tid >> 2, ac = (tid & 3) * 8;

  for (int t = 0; t < T_; ++t) {
    const bf16* h_in = (t & 1) ? hbu1 : hbu0;
    bf16* h_out      = (t & 1) ? hbu0 : hbu1;

    v8f acc[4][2];
    for (int g = 0; g < 4; ++g)
      for (int j = 0; j < 2; ++j) acc[g][j] = (v8f)(0.0f);

    // issue chunk 0 (5 async ops per wave: 1 for A, 4 for B)
    async_ld_b128(&As0[ar * LDT + ac], h_in + (size_t)ar * H_ + ac);
    for (int i = tid; i < 1024; i += 256) {
      int r = i >> 2, c = (i & 3) * 8;
      int gate = r >> 6, col = r & 63;
      async_ld_b128(&Bs0[r * LDT + c],
                    Whh + (size_t)(gate * H_ + hc0 + col) * H_ + c);
    }

    for (int kc = 0; kc < NK; ++kc) {
      int cur = kc & 1;
      bf16* Acur = cur ? As1 : As0;
      bf16* Bcur = cur ? Bs1 : Bs0;
      bf16* Anxt = cur ? As0 : As1;
      bf16* Bnxt = cur ? Bs0 : Bs1;
      if (kc + 1 < NK) {
        int k0 = (kc + 1) * KC;            // prefetch next chunk
        async_ld_b128(&Anxt[ar * LDT + ac], h_in + (size_t)ar * H_ + k0 + ac);
        for (int i = tid; i < 1024; i += 256) {
          int r = i >> 2, c = (i & 3) * 8;
          int gate = r >> 6, col = r & 63;
          async_ld_b128(&Bnxt[r * LDT + c],
                        Whh + (size_t)(gate * H_ + hc0 + col) * H_ + k0 + c);
        }
        wait_async_le5();                  // chunk kc retired (in-order cnt)
      } else {
        wait_async_0();
      }
      __syncthreads();
      v16bf a = frag_ld(&Acur[rt * 16 * LDT]);
      for (int g = 0; g < 4; ++g)
        for (int j = 0; j < 2; ++j) {
          v16bf b = frag_ld(&Bcur[(g * 64 + cth + j * 16) * LDT]);
          acc[g][j] = wmma_bf16(a, b, acc[g][j]);
        }
      __syncthreads();                     // all waves done reading cur buf
    }

    // fused LSTM cell epilogue
    for (int j = 0; j < 2; ++j) {
      int n = hc0 + cth + j * 16 + cn;
      for (int r = 0; r < 8; ++r) {
        int m = rt * 16 + mo + r;          // batch row 0..63
        size_t xb = ((size_t)t * B_ + m) * (size_t)G_;
        float iv = acc[0][j][r] + (float)xg[xb + 0 * H_ + n];
        float fv = acc[1][j][r] + (float)xg[xb + 1 * H_ + n];
        float gv = acc[2][j][r] + (float)xg[xb + 2 * H_ + n];
        float ov = acc[3][j][r] + (float)xg[xb + 3 * H_ + n];
        float cnew = sigmoidf_(fv) * creg[j][r] + sigmoidf_(iv) * tanhf(gv);
        float hnew = sigmoidf_(ov) * tanhf(cnew);
        float mv = mask[t * B_ + m];
        hnew *= mv;
        cnew *= mv;
        creg[j][r] = cnew;
        h_out[(size_t)m * H_ + n] = (bf16)hnew;
        latent[((size_t)t * B_ + m) * H_ + n] = (bf16)hnew;
        if (t == T_ - 1) {
          outH[(size_t)m * H_ + n] = hnew;
          outC[(size_t)m * H_ + n] = cnew;
        }
      }
    }

    // device-scope barrier between steps (monotonic counter)
    __threadfence();                       // release this thread's h_out stores
    __syncthreads();
    if (tid == 0) {
      __hip_atomic_fetch_add(bar, 1u, __ATOMIC_ACQ_REL, __HIP_MEMORY_SCOPE_AGENT);
      unsigned target = (unsigned)(t + 1) * gridDim.x;
      while (__hip_atomic_load(bar, __ATOMIC_ACQUIRE, __HIP_MEMORY_SCOPE_AGENT) <
             target) {
        __builtin_amdgcn_s_sleep(1);
      }
      __threadfence();
    }
    __syncthreads();
  }
}

// ---------------------------------------------------------------------------
// decoder: values[TB,18] = (latent @ Wdec^T + bdec) * mask  (N padded to 32)
// ---------------------------------------------------------------------------
__global__ __launch_bounds__(256) void dec_kernel(
    const bf16* __restrict__ latent, const float* __restrict__ Wdec,
    const float* __restrict__ bdec, const float* __restrict__ mask,
    float* __restrict__ values) {
  __shared__ __align__(16) bf16 As[128 * LDT];
  __shared__ __align__(16) bf16 Bs[32 * LDT];
  int tid  = threadIdx.x;
  int wave = tid >> 5;
  int row0 = blockIdx.x * 128;

  v8f acc[2];
  acc[0] = (v8f)(0.0f);
  acc[1] = (v8f)(0.0f);

  for (int k0 = 0; k0 < H_; k0 += KC) {
    __syncthreads();
    for (int i = tid; i < 128 * (KC / 8); i += 256) {   // A bf16
      int r = i >> 2, c = (i & 3) * 8;
      *(bf16x8*)(&As[r * LDT + c]) =
          *(const bf16x8*)(latent + (size_t)(row0 + r) * H_ + k0 + c);
    }
    {                                                   // B: 18 rows + zero pad
      int i = tid;                                      // 32*(KC/4)=256 exactly
      int r = i >> 3, c = (i & 7) * 4;
      bf16x4 b = { (bf16)0.0f, (bf16)0.0f, (bf16)0.0f, (bf16)0.0f };
      if (r < A_) {
        float4 v = *(const float4*)(Wdec + (size_t)r * H_ + k0 + c);
        b = bf16x4{ (bf16)v.x, (bf16)v.y, (bf16)v.z, (bf16)v.w };
      }
      *(bf16x4*)(&Bs[r * LDT + c]) = b;
    }
    __syncthreads();
    v16bf a  = frag_ld(&As[wave * 16 * LDT]);
    v16bf b0 = frag_ld(&Bs[0]);
    v16bf b1 = frag_ld(&Bs[16 * LDT]);
    acc[0] = wmma_bf16(a, b0, acc[0]);
    acc[1] = wmma_bf16(a, b1, acc[1]);
  }

  int lane = tid & 31, cn = lane & 15, mo = (lane >> 4) * 8;
  for (int ct = 0; ct < 2; ++ct) {
    int n = ct * 16 + cn;
    if (n < A_) {
      float bias = bdec[n];
      for (int r = 0; r < 8; ++r) {
        int m = row0 + wave * 16 + mo + r;      // flattened (t,b)
        float v = (acc[ct][r] + bias) * mask[m];
        values[(size_t)m * A_ + n] = v;
      }
    }
  }
}

// ---------------------------------------------------------------------------
extern "C" void kernel_launch(void* const* d_in, const int* in_sizes, int n_in,
                              void* d_out, int out_size, void* d_ws,
                              size_t ws_size, hipStream_t stream) {
  const float* obs  = (const float*)d_in[0];
  const float* mask = (const float*)d_in[1];
  const float* Wenc = (const float*)d_in[2];
  const float* benc = (const float*)d_in[3];
  const float* Wih  = (const float*)d_in[4];
  const float* bih  = (const float*)d_in[5];
  const float* Whh  = (const float*)d_in[6];
  const float* bhh  = (const float*)d_in[7];
  const float* Wdec = (const float*)d_in[8];
  const float* bdec = (const float*)d_in[9];

  float* outV = (float*)d_out;                         // T*B*A
  float* outH = outV + (size_t)TB_ * A_;               // B*H
  float* outC = outH + (size_t)B_ * H_;                // B*H

  char* ws = (char*)d_ws;
  bf16* xg   = (bf16*)ws;                              // TB*4H bf16 (256MB)
  bf16* enc  = xg  + (size_t)TB_ * G_;                 // TB*H bf16 (64MB)
  bf16* lat  = enc + (size_t)TB_ * H_;                 // TB*H bf16 (64MB)
  bf16* hbu0 = lat + (size_t)TB_ * H_;                 // B*H bf16
  bf16* hbu1 = hbu0 + (size_t)B_ * H_;                 // B*H bf16
  bf16* whhb = hbu1 + (size_t)B_ * H_;                 // 4H*H bf16 (8MB)
  unsigned* bar = (unsigned*)(whhb + (size_t)G_ * H_); // barrier counter

  init_kernel<<<(B_ * H_ + 255) / 256, 256, 0, stream>>>(hbu0, bar);
  cvt_whh_kernel<<<(G_ * H_ / 4 + 255) / 256, 256, 0, stream>>>(Whh, whhb);
  enc_kernel<<<dim3(TB_ / 128, H_ / 128), 256, 0, stream>>>(obs, Wenc, benc, enc);
  xg_kernel<<<dim3(TB_ / 128, G_ / 128), 256, 0, stream>>>(enc, Wih, bih, bhh, xg);

  lstm_scan_kernel<<<H_ / 64, 256, 0, stream>>>(hbu0, hbu1, xg, whhb, mask, lat,
                                                outH, outC, bar);

  dec_kernel<<<TB_ / 128, 256, 0, stream>>>(lat, Wdec, bdec, mask, outV);
}
